// ParallelDeepseekV2Attention_28492813042031
// MI455X (gfx1250) — matrix operand order repaired and checked by reading, not compiled, not run on hardware
//
#include <hip/hip_runtime.h>
#include <math.h>
#include <stdint.h>

// ---------------------------------------------------------------------------
// DeepSeek-V2 MLA attention forward for gfx1250 (wave32, WMMA f16 -> f32)
// ---------------------------------------------------------------------------
#define SEQLEN 2048
#define NHEADS 16
#define QHEAD  192
#define DNOPE  128
#define DROPE  64
#define DV     128

typedef __attribute__((ext_vector_type(16))) _Float16 v16h;
typedef __attribute__((ext_vector_type(8)))  _Float16 v8h;
typedef __attribute__((ext_vector_type(8)))  float    v8f;

#define WMMA16(a, b, c) \
  __builtin_amdgcn_wmma_f32_16x16x32_f16(false, (a), false, (b), (short)0, (c), false, false)

__device__ __forceinline__ v8f vzero8() {
  v8f z;
#pragma unroll
  for (int i = 0; i < 8; ++i) z[i] = 0.0f;
  return z;
}

// A fragment (16x32 f16): lane group g = lane>>4 holds K chunks {8g..8g+7, 16+8g..23+8g}
__device__ __forceinline__ v16h frag_a(const _Float16* p, int lane) {
  const int klo = (lane >> 4) * 8;
  v8h lo = *(const v8h*)(p + klo);
  v8h hi = *(const v8h*)(p + klo + 16);
  v16h r;
#pragma unroll
  for (int i = 0; i < 8; ++i) { r[i] = lo[i]; r[8 + i] = hi[i]; }
  return r;
}

// B fragment (32x16 f16): lane group g holds K chunks {8g..8g+7, 16+8g..23+8g}; lane%16 = column
__device__ __forceinline__ v16h frag_b(const _Float16* p, int lane) {
  const int g = (lane >> 4) * 8;
  v8h lo = *(const v8h*)(p + g);
  v8h hi = *(const v8h*)(p + g + 16);
  v16h r;
#pragma unroll
  for (int i = 0; i < 8; ++i) { r[i] = lo[i]; r[8 + i] = hi[i]; }
  return r;
}

// Async global->LDS copy of 16 bytes per lane (CDNA5, ASYNCcnt-tracked)
__device__ __forceinline__ void async_ld16(const _Float16* gsrc, _Float16* ldst) {
  const unsigned lds_off = (unsigned)(uintptr_t)ldst;  // low 32 bits = LDS offset
  asm volatile("global_load_async_to_lds_b128 %0, %1, off"
               :: "v"(lds_off), "v"(gsrc)
               : "memory");
}
__device__ __forceinline__ void wait_async0() {
  asm volatile("s_wait_asynccnt 0x0" ::: "memory");
}
__device__ __forceinline__ void wait_ds0() {
  asm volatile("s_wait_dscnt 0x0" ::: "memory");
}

// ---------------------------------------------------------------------------
// WMMA GEMM: C(MxN) = A(MxK) * B(KxN).  A: float or _Float16, B: float.
// Block = 256 threads = 8 waves (4x2), macro tile 128xBN, K step 32,
// register-staged double buffering (global loads overlap WMMA).
// ---------------------------------------------------------------------------
template <typename AT, typename OT, int BN>
__global__ __launch_bounds__(256) void gemm_wmma(const AT* __restrict__ A,
                                                 const float* __restrict__ B,
                                                 OT* __restrict__ C,
                                                 int M, int N, int K) {
  constexpr int NFRAG = BN / 32;     // B fragments per wave (2 or 4)
  __shared__ _Float16 As[128][40];   // row stride 80B (16B aligned)
  __shared__ _Float16 Bt[BN][40];    // B tile transposed: [n][k]
  const int tid  = threadIdx.x;
  const int lane = tid & 31;
  const int wid  = tid >> 5;
  const int m0 = blockIdx.y * 128;
  const int n0 = blockIdx.x * BN;
  const int wm = (wid & 3) * 32;
  const int wn = (wid >> 2) * (BN / 2);
  const int rl = lane & 15;
  (void)M;

  v8f acc[2][NFRAG];
#pragma unroll
  for (int i = 0; i < 2; ++i)
#pragma unroll
    for (int j = 0; j < NFRAG; ++j) acc[i][j] = vzero8();

  // register staging buffers for the double buffer
  float4 aF[4];
  v8h    aH[2];
  float4 bF[NFRAG];

  auto ldA = [&](int k0) {
    if constexpr (sizeof(AT) == 4) {
#pragma unroll
      for (int t = 0; t < 4; ++t) {            // 128x32 f32 = 1024 float4
        const int i4 = tid + t * 256;
        const int r = i4 >> 3, c4 = i4 & 7;
        aF[t] = *(const float4*)&A[(size_t)(m0 + r) * K + k0 + c4 * 4];
      }
    } else {
#pragma unroll
      for (int t = 0; t < 2; ++t) {            // 128x32 f16 = 512 v8h
        const int i8 = tid + t * 256;
        const int r = i8 >> 2, c8 = i8 & 3;
        aH[t] = *(const v8h*)&A[(size_t)(m0 + r) * K + k0 + c8 * 8];
      }
    }
  };
  auto ldB = [&](int k0) {
#pragma unroll
    for (int t = 0; t < NFRAG; ++t) {          // 32xBN f32 = 8*BN float4
      const int i4 = tid + t * 256;
      const int r = i4 / (BN / 4), c4 = i4 % (BN / 4);
      bF[t] = *(const float4*)&B[(size_t)(k0 + r) * N + n0 + c4 * 4];
    }
  };
  auto stA = [&]() {
    if constexpr (sizeof(AT) == 4) {
#pragma unroll
      for (int t = 0; t < 4; ++t) {
        const int i4 = tid + t * 256;
        const int r = i4 >> 3, c4 = i4 & 7;
        As[r][c4 * 4 + 0] = (_Float16)aF[t].x;
        As[r][c4 * 4 + 1] = (_Float16)aF[t].y;
        As[r][c4 * 4 + 2] = (_Float16)aF[t].z;
        As[r][c4 * 4 + 3] = (_Float16)aF[t].w;
      }
    } else {
#pragma unroll
      for (int t = 0; t < 2; ++t) {
        const int i8 = tid + t * 256;
        const int r = i8 >> 2, c8 = i8 & 3;
        *(v8h*)&As[r][c8 * 8] = aH[t];
      }
    }
  };
  auto stB = [&]() {
#pragma unroll
    for (int t = 0; t < NFRAG; ++t) {
      const int i4 = tid + t * 256;
      const int r = i4 / (BN / 4), c4 = i4 % (BN / 4);
      Bt[c4 * 4 + 0][r] = (_Float16)bF[t].x;
      Bt[c4 * 4 + 1][r] = (_Float16)bF[t].y;
      Bt[c4 * 4 + 2][r] = (_Float16)bF[t].z;
      Bt[c4 * 4 + 3][r] = (_Float16)bF[t].w;
    }
  };

  ldA(0); ldB(0);
  stA(); stB();
  __syncthreads();

  for (int k0 = 0; k0 < K; k0 += 32) {
    const bool more = (k0 + 32) < K;
    if (more) { ldA(k0 + 32); ldB(k0 + 32); }  // overlap next global loads with WMMA
    v16h a0 = frag_a(&As[wm + rl][0], lane);
    v16h a1 = frag_a(&As[wm + 16 + rl][0], lane);
#pragma unroll
    for (int j = 0; j < NFRAG; ++j) {
      v16h bf = frag_b(&Bt[wn + 16 * j + rl][0], lane);
      acc[0][j] = WMMA16(a0, bf, acc[0][j]);
      acc[1][j] = WMMA16(a1, bf, acc[1][j]);
    }
    __syncthreads();
    if (more) { stA(); stB(); __syncthreads(); }
  }

  // C layout: lane%16 = column, row = vgpr + 8*(lane>>4)
#pragma unroll
  for (int im = 0; im < 2; ++im)
#pragma unroll
    for (int j = 0; j < NFRAG; ++j) {
      const int mb = m0 + wm + im * 16 + (lane >> 4) * 8;
      const int nb = n0 + wn + 16 * j + rl;
#pragma unroll
      for (int r = 0; r < 8; ++r)
        C[(size_t)(mb + r) * N + nb] = (OT)acc[im][j][r];
    }
}

// ---------------------------------------------------------------------------
// RMS norm kernels (fp32 math), one block per row
// ---------------------------------------------------------------------------
__global__ __launch_bounds__(256) void rms_rows_inplace(float* __restrict__ x,
                                                        const float* __restrict__ w,
                                                        int n) {
  __shared__ float red[256];
  float* p = x + (size_t)blockIdx.x * n;
  float ss = 0.0f;
  for (int c = threadIdx.x; c < n; c += 256) { const float v = p[c]; ss += v * v; }
  red[threadIdx.x] = ss;
  __syncthreads();
  for (int s = 128; s > 0; s >>= 1) {
    if (threadIdx.x < s) red[threadIdx.x] += red[threadIdx.x + s];
    __syncthreads();
  }
  const float inv = rsqrtf(red[0] / (float)n + 1e-6f);
  for (int c = threadIdx.x; c < n; c += 256) p[c] = p[c] * inv * w[c];
}

__global__ __launch_bounds__(256) void rms_kv_to_f16(const float* __restrict__ ckv,
                                                     const float* __restrict__ w,
                                                     _Float16* __restrict__ outn) {
  __shared__ float red[256];
  const float* p = ckv + (size_t)blockIdx.x * 576;
  float ss = 0.0f;
  for (int c = threadIdx.x; c < 512; c += 256) { const float v = p[c]; ss += v * v; }
  red[threadIdx.x] = ss;
  __syncthreads();
  for (int s = 128; s > 0; s >>= 1) {
    if (threadIdx.x < s) red[threadIdx.x] += red[threadIdx.x + s];
    __syncthreads();
  }
  const float inv = rsqrtf(red[0] / 512.0f + 1e-6f);
  for (int c = threadIdx.x; c < 512; c += 256)
    outn[(size_t)blockIdx.x * 512 + c] = (_Float16)(p[c] * inv * w[c]);
}

// ---------------------------------------------------------------------------
// YaRN RoPE (deinterleaved), cos/sin computed in double per (pos, freq)
// ---------------------------------------------------------------------------
__device__ __forceinline__ void yarn_cos_sin(int pos, int j, float* cc, float* ss) {
  const double LNB   = 9.210340371976184;   // ln(10000)
  const double TWOPI = 6.283185307179586;
  const double fx = (double)j / 32.0;
  const double fe = exp(-fx * LNB);         // extrapolation freq
  const double fi = fe / 40.0;              // interpolation freq (yarn factor 40)
  const double low  = floor(64.0 * log(4096.0 / (32.0 * TWOPI)) / (2.0 * LNB));  // 10
  const double high = ceil (64.0 * log(4096.0 / ( 1.0 * TWOPI)) / (2.0 * LNB));  // 23
  double ramp = ((double)j - low) / (high - low);
  ramp = fmin(fmax(ramp, 0.0), 1.0);
  const double mask = 1.0 - ramp;
  const double invf = fi * (1.0 - mask) + fe * mask;
  const double ang = (double)pos * invf;
  *cc = (float)cos(ang);   // _m == 1 since MSCALE == MSCALE_ALL_DIM
  *ss = (float)sin(ang);
}

// q: (B*S, NHEADS, 192) f16, rope applied in place on dims [128,192)
__global__ __launch_bounds__(512) void rope_q(_Float16* __restrict__ q) {
  const int idx = blockIdx.x;
  const int pos = idx & (SEQLEN - 1);
  const int t = threadIdx.x;
  const int h = t >> 5;
  const int j = t & 31;
  float c, s;
  yarn_cos_sin(pos, j, &c, &s);
  const size_t base = ((size_t)idx * NHEADS + h) * QHEAD + DNOPE;
  const float xe = (float)q[base + 2 * j];
  const float xo = (float)q[base + 2 * j + 1];
  __syncthreads();  // deinterleave permutes within the 64-dim block
  q[base + j]      = (_Float16)(xe * c - xo * s);
  q[base + 32 + j] = (_Float16)(xo * c + xe * s);
}

// ckv fp32 (B*S, 576) -> roped k_pe f16 (B*S, 64)
__global__ __launch_bounds__(32) void rope_k(const float* __restrict__ ckv,
                                             _Float16* __restrict__ kpe) {
  const int idx = blockIdx.x;
  const int pos = idx & (SEQLEN - 1);
  const int j = threadIdx.x;
  float c, s;
  yarn_cos_sin(pos, j, &c, &s);
  const float xe = ckv[(size_t)idx * 576 + 512 + 2 * j];
  const float xo = ckv[(size_t)idx * 576 + 512 + 2 * j + 1];
  kpe[(size_t)idx * 64 + j]      = (_Float16)(xe * c - xo * s);
  kpe[(size_t)idx * 64 + 32 + j] = (_Float16)(xo * c + xe * s);
}

// ---------------------------------------------------------------------------
// Flash attention: one wave per (b, h, 16-row q tile); 64-key k tiles, causal.
// Q/K tiles fetched with async global->LDS DMA; V transposed through VGPRs.
// q:(B*S,NH,192)f16  kv:(B*S,NH,256)f16 [k_nope|v]  kpe:(B*S,64)f16
// out:(B*S, NH*128)f16
// ---------------------------------------------------------------------------
__global__ __launch_bounds__(32) void attn_kernel(const _Float16* __restrict__ q,
                                                  const _Float16* __restrict__ kv,
                                                  const _Float16* __restrict__ kpe,
                                                  _Float16* __restrict__ out) {
  const int qt = blockIdx.x;      // q tile (16 rows)
  const int h  = blockIdx.y;
  const int b  = blockIdx.z;
  const int lane = threadIdx.x;
  const int q0 = qt * 16;
  const size_t bS = (size_t)b * SEQLEN;
  const int rl  = lane & 15;
  const int grp = lane >> 4;

  __shared__ _Float16 Qs[16][200];   // 16 x 192 (pad keeps 16B chunks aligned)
  __shared__ _Float16 Ks[64][200];   // 64 keys x 192
  __shared__ _Float16 Vt[128][72];   // V transposed: [v-dim][key]
  __shared__ float    Sf[16][64];
  __shared__ _Float16 Pp[16][72];
  __shared__ float    row_corr[16];
  __shared__ float    row_sum_sh[16];

  const float MSC   = 1.0f + 0.1f * 0.707f * 3.6888794541139363f;  // yarn mscale(40,.707)
  const float SCALE = (MSC * MSC) * rsqrtf((float)QHEAD);

  // Q tile via async DMA (waited together with the first K tile)
  for (int i = lane; i < 16 * 24; i += 32) {
    const int r = i / 24, c = (i % 24) * 8;
    async_ld16(&q[((bS + q0 + r) * NHEADS + h) * QHEAD + c], &Qs[r][c]);
  }

  v8f o[8];
#pragma unroll
  for (int t = 0; t < 8; ++t) o[t] = vzero8();
  float m_run = -INFINITY;
  float l_run = 0.0f;

  const int ktmax = (q0 + 15) >> 6;
  for (int kt = 0; kt <= ktmax; ++kt) {
    const int k0 = kt * 64;
    wait_ds0();  // all LDS reads of the previous tile done before DMA overwrites it
    // K tile via async DMA: 128 nope dims from kv, 64 rope dims from kpe
    for (int i = lane; i < 64 * 24; i += 32) {
      const int r = i / 24, c8 = i % 24;
      if (c8 < 16)
        async_ld16(&kv[((bS + k0 + r) * NHEADS + h) * 256 + c8 * 8], &Ks[r][c8 * 8]);
      else
        async_ld16(&kpe[(bS + k0 + r) * 64 + (c8 - 16) * 8], &Ks[r][c8 * 8]);
    }
    // V tile transposed (manual; overlaps with the async DMA above)
    for (int i = lane; i < 64 * 16; i += 32) {
      const int r = i / 16, c8 = i % 16;
      v8h v = *(const v8h*)&kv[((bS + k0 + r) * NHEADS + h) * 256 + 128 + c8 * 8];
#pragma unroll
      for (int jj = 0; jj < 8; ++jj) Vt[c8 * 8 + jj][r] = v[jj];
    }
    wait_async0();  // Q + K tiles resident in LDS
    // S = Q K^T  (16x64), contraction over 192 in 6 slices of 32
    for (int nt = 0; nt < 4; ++nt) {
      v8f acc = vzero8();
#pragma unroll
      for (int ks = 0; ks < 6; ++ks) {
        v16h a  = frag_a(&Qs[rl][ks * 32], lane);
        v16h bb = frag_b(&Ks[nt * 16 + rl][ks * 32], lane);
        acc = WMMA16(a, bb, acc);
      }
#pragma unroll
      for (int r = 0; r < 8; ++r)
        Sf[r + grp * 8][nt * 16 + rl] = acc[r];
    }
    // online softmax (lanes 0..15, one row each; single wave => in-order LDS)
    if (lane < 16) {
      const int qg = q0 + lane;
      float tm = -INFINITY;
      for (int c = 0; c < 64; ++c) {
        float sv = Sf[lane][c] * SCALE;
        if (k0 + c > qg) sv = -INFINITY;  // causal mask
        Sf[lane][c] = sv;
        tm = fmaxf(tm, sv);
      }
      const float mnew = fmaxf(m_run, tm);
      const float corr = __expf(m_run - mnew);  // 0 on first tile
      float psum = 0.0f;
      for (int c = 0; c < 64; ++c) {
        const float p = __expf(Sf[lane][c] - mnew);
        Pp[lane][c] = (_Float16)p;
        psum += p;
      }
      l_run = l_run * corr + psum;
      m_run = mnew;
      row_corr[lane] = corr;
    }
    float cr[8];
#pragma unroll
    for (int r = 0; r < 8; ++r) cr[r] = row_corr[r + grp * 8];
#pragma unroll
    for (int t = 0; t < 8; ++t)
#pragma unroll
      for (int r = 0; r < 8; ++r) o[t][r] *= cr[r];
    // O += P V  (16x128), contraction over 64 keys in 2 slices of 32
    for (int nt = 0; nt < 8; ++nt) {
#pragma unroll
      for (int ks = 0; ks < 2; ++ks) {
        v16h a  = frag_a(&Pp[rl][ks * 32], lane);
        v16h bb = frag_b(&Vt[nt * 16 + rl][ks * 32], lane);
        o[nt] = WMMA16(a, bb, o[nt]);
      }
    }
  }
  if (lane < 16) row_sum_sh[lane] = l_run;
  float ir[8];
#pragma unroll
  for (int r = 0; r < 8; ++r) ir[r] = 1.0f / row_sum_sh[r + grp * 8];
#pragma unroll
  for (int nt = 0; nt < 8; ++nt)
#pragma unroll
    for (int r = 0; r < 8; ++r)
      out[(bS + q0 + grp * 8 + r) * (NHEADS * DV) + h * DV + nt * 16 + rl] =
          (_Float16)(o[nt][r] * ir[r]);
}

// ---------------------------------------------------------------------------
// Host-side orchestration
// ---------------------------------------------------------------------------
extern "C" void kernel_launch(void* const* d_in, const int* in_sizes, int n_in,
                              void* d_out, int out_size, void* d_ws, size_t ws_size,
                              hipStream_t stream) {
  (void)in_sizes; (void)n_in; (void)out_size; (void)ws_size;
  const float* hidden    = (const float*)d_in[0];
  // d_in[1] = position_ids (arange; recomputed in-kernel)
  const float* wq_a      = (const float*)d_in[2];
  const float* q_norm_w  = (const float*)d_in[3];
  const float* wq_b      = (const float*)d_in[4];
  const float* wkv_a     = (const float*)d_in[5];
  const float* kv_norm_w = (const float*)d_in[6];
  const float* wkv_b     = (const float*)d_in[7];
  const float* wo        = (const float*)d_in[8];
  float* out = (float*)d_out;
  char*  ws  = (char*)d_ws;

  // workspace layout (bytes), R = b*s = 4096 rows
  float*    qa   = (float*)   (ws + 0);                 // R x 1536 f32   (25.2 MB)
  _Float16* qh   = (_Float16*)(ws + 25165824);          // R x 3072 f16   (25.2 MB)
  float*    ckv  = (float*)   (ws + 50331648);          // R x 576  f32   ( 9.4 MB)
  _Float16* ckvn = (_Float16*)(ws + 59768832);          // R x 512  f16   ( 4.2 MB)
  _Float16* kvh  = (_Float16*)(ws + 63963136);          // R x 4096 f16   (33.6 MB)
  _Float16* kpe  = (_Float16*)(ws + 97517568);          // R x 64   f16   ( 0.5 MB)
  _Float16* att  = (_Float16*)(ws + 98041856);          // R x 2048 f16   (16.8 MB)

  // q_a = hidden @ wq_a              (4096 x 1536, K=2048)
  gemm_wmma<float, float, 128><<<dim3(1536 / 128, 32), 256, 0, stream>>>(
      hidden, wq_a, qa, 4096, 1536, 2048);
  // rms_norm(q_a) in place
  rms_rows_inplace<<<4096, 256, 0, stream>>>(qa, q_norm_w, 1536);
  // q = q_a_n @ wq_b  -> f16         (4096 x 3072, K=1536)
  gemm_wmma<float, _Float16, 128><<<dim3(3072 / 128, 32), 256, 0, stream>>>(
      qa, wq_b, qh, 4096, 3072, 1536);
  // ckv = hidden @ wkv_a             (4096 x 576, K=2048) - N not /128, use BN=64
  gemm_wmma<float, float, 64><<<dim3(576 / 64, 32), 256, 0, stream>>>(
      hidden, wkv_a, ckv, 4096, 576, 2048);
  // rms_norm(compressed_kv) -> f16
  rms_kv_to_f16<<<4096, 256, 0, stream>>>(ckv, kv_norm_w, ckvn);
  // kv = ckv_n @ wkv_b -> f16        (4096 x 4096, K=512)
  gemm_wmma<_Float16, _Float16, 128><<<dim3(4096 / 128, 32), 256, 0, stream>>>(
      ckvn, wkv_b, kvh, 4096, 4096, 512);
  // YaRN rope
  rope_q<<<4096, 512, 0, stream>>>(qh);
  rope_k<<<4096, 32, 0, stream>>>(ckv, kpe);
  // flash attention
  attn_kernel<<<dim3(SEQLEN / 16, NHEADS, 2), 32, 0, stream>>>(qh, kvh, kpe, att);
  // out = attn @ wo                  (4096 x 2048, K=2048)
  gemm_wmma<_Float16, float, 128><<<dim3(2048 / 128, 32), 256, 0, stream>>>(
      att, wo, out, 4096, 2048, 2048);
}
